// CausalLocalSelfAttention_84293028151817
// MI455X (gfx1250) — matrix operand order repaired and checked by reading, not compiled, not run on hardware
//
#include <hip/hip_runtime.h>

typedef _Float16 half_t;
typedef __attribute__((ext_vector_type(16))) _Float16 v16h;
typedef __attribute__((ext_vector_type(8)))  _Float16 v8h;
typedef __attribute__((ext_vector_type(8)))  float    v8f;

#define WMMA_F16(a, b, c) \
  __builtin_amdgcn_wmma_f32_16x16x32_f16(false, (a), false, (b), (short)0, (c), false, false)

static constexpr int B_ = 2, T_ = 2048, C_ = 1024, H_ = 16, HD = 64;
static constexpr int M_ = B_ * T_;  // 4096 flattened rows
static constexpr float NEG_INF_F = -1e30f;

// ---------------------------------------------------------------------------
// f32 -> f16 conversion
// ---------------------------------------------------------------------------
__global__ void cvt_f32_to_f16(const float* __restrict__ src, half_t* __restrict__ dst, int n) {
  int i = blockIdx.x * blockDim.x + threadIdx.x;
  if (i < n) dst[i] = (half_t)src[i];
}

// ---------------------------------------------------------------------------
// Fragment loaders (wave32; this code's contiguous k order, same for A and B)
// A 16x32 f16: lane row = lane%16; half i -> k = 16*(lane/16) + i (contiguous)
// B 32x16 f16: lane col = lane%16; half i -> k = 16*(lane/16) + i (contiguous)
// ---------------------------------------------------------------------------
__device__ static inline v16h load_a_g(const half_t* base, int ld, int lane) {
  const half_t* p = base + (size_t)(lane & 15) * ld + ((lane >> 4) << 4);
  v8h lo = *(const v8h*)p;
  v8h hi = *(const v8h*)(p + 8);
  v16h r;
#pragma unroll
  for (int i = 0; i < 8; ++i) { r[i] = lo[i]; r[i + 8] = hi[i]; }
  return r;
}
__device__ static inline v16h load_b_g(const half_t* base, int ld, int lane) {
  return *(const v16h*)(base + (size_t)(lane & 15) * ld + ((lane >> 4) << 4));
}

// ---------------------------------------------------------------------------
// Double-buffered 32x64 GEMM core (NT, K = C_ = 1024).
// acc[j*2+rt]: j = 16-col sub-tile (0..3), rt = 16-row sub-tile (0..1).
// Next step's fragments are loaded BEFORE the current step's 8 WMMAs so the
// scheduler overlaps global loads with matrix work.
// ---------------------------------------------------------------------------
__device__ static inline void gemm_core_32x64(const half_t* __restrict__ abase,
                                              const half_t* __restrict__ bbase,
                                              v8f acc[8], int lane) {
  v16h a0 = load_a_g(abase, C_, lane);
  v16h a1 = load_a_g(abase + (size_t)16 * C_, C_, lane);
  v16h b0 = load_b_g(bbase + (size_t)0 * C_, C_, lane);
  v16h b1 = load_b_g(bbase + (size_t)16 * C_, C_, lane);
  v16h b2 = load_b_g(bbase + (size_t)32 * C_, C_, lane);
  v16h b3 = load_b_g(bbase + (size_t)48 * C_, C_, lane);
  for (int kk = 0; kk < C_; kk += 32) {
    const int kn = (kk + 32 < C_) ? kk + 32 : kk;  // last iter: redundant, in-bounds
    v16h an0 = load_a_g(abase + kn, C_, lane);
    v16h an1 = load_a_g(abase + (size_t)16 * C_ + kn, C_, lane);
    v16h bn0 = load_b_g(bbase + (size_t)0 * C_ + kn, C_, lane);
    v16h bn1 = load_b_g(bbase + (size_t)16 * C_ + kn, C_, lane);
    v16h bn2 = load_b_g(bbase + (size_t)32 * C_ + kn, C_, lane);
    v16h bn3 = load_b_g(bbase + (size_t)48 * C_ + kn, C_, lane);
    acc[0] = WMMA_F16(a0, b0, acc[0]);
    acc[1] = WMMA_F16(a1, b0, acc[1]);
    acc[2] = WMMA_F16(a0, b1, acc[2]);
    acc[3] = WMMA_F16(a1, b1, acc[3]);
    acc[4] = WMMA_F16(a0, b2, acc[4]);
    acc[5] = WMMA_F16(a1, b2, acc[5]);
    acc[6] = WMMA_F16(a0, b3, acc[6]);
    acc[7] = WMMA_F16(a1, b3, acc[7]);
    a0 = an0; a1 = an1; b0 = bn0; b1 = bn1; b2 = bn2; b3 = bn3;
  }
}

// ---------------------------------------------------------------------------
// Kernel 1: qkv = x @ w_qkv^T (M=4096, N=3072, K=1024). 32x64 per wave.
// Writes q (scaled by hd^-0.5) and k as (B,H,T,hd) f16, v transposed (B,H,hd,T)
// ---------------------------------------------------------------------------
__global__ __launch_bounds__(256) void qkv_gemm(const half_t* __restrict__ xf,
                                                const half_t* __restrict__ wf,
                                                half_t* __restrict__ qf,
                                                half_t* __restrict__ kf,
                                                half_t* __restrict__ vtf) {
  const int lane = threadIdx.x & 31;
  const int wid = (blockIdx.x * blockDim.x + threadIdx.x) >> 5;
  const int NC = (3 * C_) / 64;  // 48 chunks of 64 cols
  const int mt = wid / NC, nc = wid % NC;
  const int t0 = mt * 32, o0 = nc * 64;

  v8f acc[8] = {};
  gemm_core_32x64(xf + (size_t)t0 * C_, wf + (size_t)o0 * C_, acc, lane);

  const int s = o0 / C_;              // 0=q 1=k 2=v (chunk never straddles)
  const int h = (o0 % C_) / HD;
  const int bb = t0 >> 11;            // batch (32-row tile never straddles b)
  const int tt = t0 & 2047;
  const int hf = lane >> 4, n = lane & 15;
#pragma unroll
  for (int j = 0; j < 4; ++j) {
    int d = 16 * j + n;
#pragma unroll
    for (int rt = 0; rt < 2; ++rt) {
#pragma unroll
      for (int vg = 0; vg < 8; ++vg) {
        int m = 16 * rt + vg + 8 * hf;
        float val = acc[j * 2 + rt][vg];
        if (s == 0)
          qf[(((size_t)(bb * H_ + h)) * T_ + tt + m) * HD + d] = (half_t)(val * 0.125f);
        else if (s == 1)
          kf[(((size_t)(bb * H_ + h)) * T_ + tt + m) * HD + d] = (half_t)val;
        else
          vtf[(((size_t)(bb * H_ + h)) * HD + d) * T_ + tt + m] = (half_t)val;
      }
    }
  }
}

// ---------------------------------------------------------------------------
// Kernel 2: flash-style causal-local attention. One wave per 16-row q tile.
// K-block fragments double-buffered; V loads hoisted above the softmax VALU
// work so exp/shuffle hides global latency.
// ---------------------------------------------------------------------------
__global__ __launch_bounds__(128) void attn_kernel(const half_t* __restrict__ qf,
                                                   const half_t* __restrict__ kf,
                                                   const half_t* __restrict__ vtf,
                                                   const float* __restrict__ rel_bias,
                                                   half_t* __restrict__ yatt) {
  __shared__ float s_bias[257 * 16];
  __shared__ __align__(32) half_t s_P[4][16 * 32];
  for (int i = threadIdx.x; i < 257 * 16; i += blockDim.x) s_bias[i] = rel_bias[i];
  __syncthreads();

  const int lane = threadIdx.x & 31;
  const int w = threadIdx.x >> 5;
  const int wid = blockIdx.x * 4 + w;
  const int qt = wid & 127;
  const int h = (wid >> 7) & 15;
  const int bb = wid >> 11;
  const int qt0 = qt * 16;

  const half_t* qbase = qf + (((size_t)(bb * H_ + h)) * T_ + qt0) * HD;
  const half_t* kbase = kf + (((size_t)(bb * H_ + h)) * T_) * HD;
  const half_t* vbase = vtf + ((size_t)(bb * H_ + h)) * HD * T_;

  v16h qa0 = load_a_g(qbase + 0, HD, lane);
  v16h qa1 = load_a_g(qbase + 32, HD, lane);

  v8f oacc[4] = {};
  float mrow[8], lrow[8];
#pragma unroll
  for (int vg = 0; vg < 8; ++vg) { mrow[vg] = -3.0e38f; lrow[vg] = 0.f; }

  const int hf = lane >> 4, n = lane & 15;
  half_t* Pw = s_P[w];

  const int ks0 = (qt0 >= 512) ? ((qt0 - 512) & ~31) : 0;
  // preload first K block fragments
  const half_t* kbp = kbase + (size_t)ks0 * HD;
  v16h kb0 = load_b_g(kbp, HD, lane);
  v16h kb1 = load_b_g(kbp + 32, HD, lane);
  v16h kb2 = load_b_g(kbp + 16 * HD, HD, lane);
  v16h kb3 = load_b_g(kbp + 16 * HD + 32, HD, lane);

  for (int ks = ks0; ks <= qt0 + 15; ks += 32) {
    // S tile 16x32 = Q(16x64) @ K_blk^T(64x32)
    v8f sa = {}, sb = {};
    sa = WMMA_F16(qa0, kb0, sa);
    sa = WMMA_F16(qa1, kb1, sa);
    sb = WMMA_F16(qa0, kb2, sb);
    sb = WMMA_F16(qa1, kb3, sb);

    // V-block loads for this iteration (consumed after softmax)
    v16h bv0 = load_b_g(vbase + (size_t)0 * T_ + ks, T_, lane);
    v16h bv1 = load_b_g(vbase + (size_t)16 * T_ + ks, T_, lane);
    v16h bv2 = load_b_g(vbase + (size_t)32 * T_ + ks, T_, lane);
    v16h bv3 = load_b_g(vbase + (size_t)48 * T_ + ks, T_, lane);

    // prefetch next K block (clamped in-bounds; unused on last iteration)
    int ksn = ks + 32;
    if (ksn > T_ - 32) ksn = T_ - 32;
    const half_t* kbn = kbase + (size_t)ksn * HD;
    v16h nk0 = load_b_g(kbn, HD, lane);
    v16h nk1 = load_b_g(kbn + 32, HD, lane);
    v16h nk2 = load_b_g(kbn + 16 * HD, HD, lane);
    v16h nk3 = load_b_g(kbn + 16 * HD + 32, HD, lane);

    // mask + relative bias
#pragma unroll
    for (int vg = 0; vg < 8; ++vg) {
      int i = qt0 + vg + 8 * hf;
      int d1 = i - (ks + n);
      int d2 = i - (ks + 16 + n);
      float v1 = sa[vg], v2 = sb[vg];
      if (d1 < 0 || d1 > 512) v1 = NEG_INF_F;
      if (d2 < 0 || d2 > 512) v2 = NEG_INF_F;
      int c1 = min(max(d1, -128), 128) + 128;
      int c2 = min(max(d2, -128), 128) + 128;
      sa[vg] = v1 + s_bias[c1 * 16 + h];
      sb[vg] = v2 + s_bias[c2 * 16 + h];
    }

    // online softmax per row (row = vgpr + 8*(lane/16); cols across 16-lane half)
#pragma unroll
    for (int vg = 0; vg < 8; ++vg) {
      float r = fmaxf(sa[vg], sb[vg]);
      r = fmaxf(r, __shfl_xor(r, 1));
      r = fmaxf(r, __shfl_xor(r, 2));
      r = fmaxf(r, __shfl_xor(r, 4));
      r = fmaxf(r, __shfl_xor(r, 8));
      float mn = fmaxf(mrow[vg], r);
      float alpha = __expf(mrow[vg] - mn);
      mrow[vg] = mn;
      float pa = __expf(sa[vg] - mn);
      float pb = __expf(sb[vg] - mn);
      float rs = pa + pb;
      rs += __shfl_xor(rs, 1);
      rs += __shfl_xor(rs, 2);
      rs += __shfl_xor(rs, 4);
      rs += __shfl_xor(rs, 8);
      lrow[vg] = lrow[vg] * alpha + rs;
#pragma unroll
      for (int nt = 0; nt < 4; ++nt) oacc[nt][vg] *= alpha;
      int m = vg + 8 * hf;
      Pw[m * 32 + n] = (half_t)pa;
      Pw[m * 32 + 16 + n] = (half_t)pb;
    }

    // reload P from wave-private LDS tile in A-fragment layout
    v16h pA;
    {
      const half_t* pp = Pw + (lane & 15) * 32 + ((lane >> 4) << 4);
      v8h lo = *(const v8h*)pp;
      v8h hi = *(const v8h*)(pp + 8);
#pragma unroll
      for (int i2 = 0; i2 < 8; ++i2) { pA[i2] = lo[i2]; pA[i2 + 8] = hi[i2]; }
    }

    // O(16x64) += P(16x32) @ V_blk(32x64), V stored transposed (hd x T)
    oacc[0] = WMMA_F16(pA, bv0, oacc[0]);
    oacc[1] = WMMA_F16(pA, bv1, oacc[1]);
    oacc[2] = WMMA_F16(pA, bv2, oacc[2]);
    oacc[3] = WMMA_F16(pA, bv3, oacc[3]);

    kb0 = nk0; kb1 = nk1; kb2 = nk2; kb3 = nk3;
  }

  // finalize: divide by l, store (B,T,C) f16 for projection GEMM
#pragma unroll
  for (int vg = 0; vg < 8; ++vg) {
    float inv = 1.0f / lrow[vg];
    int m = vg + 8 * hf;
    size_t row = (size_t)(bb * T_ + qt0 + m);
#pragma unroll
    for (int nt = 0; nt < 4; ++nt)
      yatt[row * C_ + h * HD + nt * 16 + n] = (half_t)(oacc[nt][vg] * inv);
  }
}

// ---------------------------------------------------------------------------
// Kernel 3: y = yatt @ w_proj^T + b_proj (M=4096, N=1024, K=1024) -> f32 out
// ---------------------------------------------------------------------------
__global__ __launch_bounds__(256) void proj_gemm(const half_t* __restrict__ yf,
                                                 const half_t* __restrict__ wf,
                                                 const float* __restrict__ bias,
                                                 float* __restrict__ out) {
  const int lane = threadIdx.x & 31;
  const int wid = (blockIdx.x * blockDim.x + threadIdx.x) >> 5;
  const int NC = C_ / 64;  // 16
  const int mt = wid / NC, nc = wid % NC;
  const int t0 = mt * 32, o0 = nc * 64;

  v8f acc[8] = {};
  gemm_core_32x64(yf + (size_t)t0 * C_, wf + (size_t)o0 * C_, acc, lane);

  const int hf = lane >> 4, n = lane & 15;
#pragma unroll
  for (int j = 0; j < 4; ++j) {
    float bv = bias[o0 + 16 * j + n];
#pragma unroll
    for (int rt = 0; rt < 2; ++rt) {
#pragma unroll
      for (int vg = 0; vg < 8; ++vg) {
        int m = 16 * rt + vg + 8 * hf;
        out[(size_t)(t0 + m) * C_ + o0 + 16 * j + n] = acc[j * 2 + rt][vg] + bv;
      }
    }
  }
}

// ---------------------------------------------------------------------------
// Launch
// ---------------------------------------------------------------------------
extern "C" void kernel_launch(void* const* d_in, const int* in_sizes, int n_in,
                              void* d_out, int out_size, void* d_ws, size_t ws_size,
                              hipStream_t stream) {
  const float* x        = (const float*)d_in[0];
  const float* w_qkv    = (const float*)d_in[1];
  const float* w_proj   = (const float*)d_in[2];
  const float* b_proj   = (const float*)d_in[3];
  const float* rel_bias = (const float*)d_in[4];

  char* ws = (char*)d_ws;
  half_t* xf   = (half_t*)(ws + 0);          //  8 MB: x f16 (4096x1024)
  half_t* wqf  = (half_t*)(ws + 8388608);    //  6 MB: w_qkv f16 (3072x1024)
  half_t* wpf  = (half_t*)(ws + 14680064);   //  2 MB: w_proj f16 (1024x1024)
  half_t* qf   = (half_t*)(ws + 16777216);   //  8 MB: q f16 (B,H,T,hd), pre-scaled
  half_t* kf   = (half_t*)(ws + 25165824);   //  8 MB: k f16 (B,H,T,hd)
  half_t* vtf  = (half_t*)(ws + 33554432);   //  8 MB: v^T f16 (B,H,hd,T)
  half_t* yatt = (half_t*)(ws + 41943040);   //  8 MB: attn output f16 (B,T,C)

  const int nx = M_ * C_;
  const int nwq = 3 * C_ * C_;
  const int nwp = C_ * C_;
  cvt_f32_to_f16<<<(nx + 255) / 256, 256, 0, stream>>>(x, xf, nx);
  cvt_f32_to_f16<<<(nwq + 255) / 256, 256, 0, stream>>>(w_qkv, wqf, nwq);
  cvt_f32_to_f16<<<(nwp + 255) / 256, 256, 0, stream>>>(w_proj, wpf, nwp);

  // qkv: (4096/32) M-tiles x 48 N-chunks = 6144 waves, 8 waves/block
  qkv_gemm<<<(128 * 48) / 8, 256, 0, stream>>>(xf, wqf, qf, kf, vtf);
  // attention: 4096 q-tiles, 4 waves/block
  attn_kernel<<<4096 / 4, 128, 0, stream>>>(qf, kf, vtf, rel_bias, yatt);
  // proj: (4096/32) x 16 = 2048 waves, 8 waves/block
  proj_gemm<<<(128 * 16) / 8, 256, 0, stream>>>(yatt, wpf, b_proj, (float*)d_out);
}
